// LSTM_61555471286466
// MI455X (gfx1250) — matrix operand order repaired and checked by reading, not compiled, not run on hardware
//
#include <hip/hip_runtime.h>
#include <hip/hip_bf16.h>

// ---------------------------------------------------------------------------
// 3-layer LSTM, T=300 B=128 D=512 H=1024, fused bf16-WMMA step kernels,
// scheduled as a layer/time *wavefront*: one launch per diagonal d runs
// (layer0,t=d), (layer1,t=d-1), (layer2,t=d-2) concurrently.
//   - 302 launches instead of 900 (less launch latency on the critical path)
//   - 768 waves per launch instead of 256 (better WGP fill)
// Each wave owns a 32(M) x 16(h-col) tile across ALL FOUR gates, so the
// gating (3x sigmoid + tanh -> c,h update) happens in-register on the
// 16x16 f32 WMMA accumulators.
// ---------------------------------------------------------------------------

typedef __attribute__((ext_vector_type(16))) __bf16 v16bf;
typedef __attribute__((ext_vector_type(8)))  float  v8f;

namespace {
constexpr int T_ = 300;
constexpr int B_ = 128;
constexpr int D_ = 512;
constexpr int H_ = 1024;

union Frag {
  v16bf v;
  uint4 u[2];
};
}  // namespace

struct StepArgs {
  const __hip_bfloat16* xs[3];   // input sequence base per layer (full-t index)
  const __hip_bfloat16* w[3];    // [4,H,Din] bf16
  const __hip_bfloat16* u[3];    // [4,H,H] bf16
  const float* bias[3];          // [4,H] fp32
  float* c[3];                   // [B,H] fp32 cell state per layer
  __hip_bfloat16* hs[3];         // own-output base per layer (t & tmask index)
  float* f32;                    // layer2 fp32 output base (d_out)
  const __hip_bfloat16* h0;      // zero [B,H]
  int din[3];
  int tmask[3];                  // 0x7fffffff for full seq, 1 for ping-pong
  int d;                         // diagonal index
};

// fp32 -> bf16 elementwise convert (grid-stride)
__global__ __launch_bounds__(256) void cvt_f32_bf16(
    const float* __restrict__ in, __hip_bfloat16* __restrict__ out, long n) {
  long i = (long)blockIdx.x * blockDim.x + threadIdx.x;
  long stride = (long)gridDim.x * blockDim.x;
  for (; i < n; i += stride) out[i] = __float2bfloat16(in[i]);
}

// One wavefront diagonal: 64 blocks per layer slice, 4 waves per block.
__global__ __launch_bounds__(128) void lstm_wavefront(StepArgs a) {
  const int l = blockIdx.x >> 6;       // layer slice 0..2
  const int t = a.d - l;
  if (t < 0 || t >= T_) return;        // slice inactive on this diagonal

  const int Din = a.din[l];
  const int tm = a.tmask[l];
  const __hip_bfloat16* __restrict__ xin =
      a.xs[l] + (size_t)t * B_ * Din;
  const __hip_bfloat16* __restrict__ hprev =
      (t == 0) ? a.h0 : a.hs[l] + (size_t)((t - 1) & tm) * B_ * H_;
  __hip_bfloat16* __restrict__ hout = a.hs[l] + (size_t)(t & tm) * B_ * H_;
  float* __restrict__ houtf = (l == 2) ? a.f32 + (size_t)t * B_ * H_ : nullptr;
  const __hip_bfloat16* __restrict__ wB = a.w[l];
  const __hip_bfloat16* __restrict__ uB = a.u[l];
  const float* __restrict__ bias = a.bias[l];
  float* __restrict__ cstate = a.c[l];

  const int lane = threadIdx.x & 31;
  const int wid = (blockIdx.x & 63) * (blockDim.x >> 5) + (threadIdx.x >> 5);
  const int mbase = (wid & 3) * 32;   // batch-row tile base (2 x 16 sub-tiles)
  const int hbase = (wid >> 2) * 16;  // h-column tile base
  const int lr = lane & 15;
  const int hi = lane >> 4;           // lane half select (wave32 layout)

  v8f acc[2][4];
#pragma unroll
  for (int s = 0; s < 2; ++s)
#pragma unroll
    for (int g = 0; g < 4; ++g)
      acc[s][g] = (v8f){0.f, 0.f, 0.f, 0.f, 0.f, 0.f, 0.f, 0.f};

  // ---- input-side projection: A = x_t, B rows = W[g*H + hcol, :] ----
  {
    // A fragment (16x32 bf16): lane<16 holds K {0..7,16..23}, lane>=16 holds
    // K {8..15,24..31} of row M=lane%16 -> two b128 loads at +0 and +16 elems.
    const __hip_bfloat16* a0 = xin + (size_t)(mbase + lr) * Din + hi * 8;
    const __hip_bfloat16* a1 = a0 + (size_t)16 * Din;
    const __hip_bfloat16* pb[4];
#pragma unroll
    for (int g = 0; g < 4; ++g)
      pb[g] = wB + ((size_t)g * H_ + hbase + lr) * Din + hi * 8;

    for (int k = 0; k < Din; k += 32) {
      Frag fa0, fa1;
      fa0.u[0] = *reinterpret_cast<const uint4*>(a0 + k);
      fa0.u[1] = *reinterpret_cast<const uint4*>(a0 + k + 16);
      fa1.u[0] = *reinterpret_cast<const uint4*>(a1 + k);
      fa1.u[1] = *reinterpret_cast<const uint4*>(a1 + k + 16);
#pragma unroll
      for (int g = 0; g < 4; ++g) {
        // B fragment (32x16 bf16): lane holds column lane%16,
        // K {0..7,16..23} (lane<16) or {8..15,24..31} (lane>=16), same as A.
        Frag fb;
        fb.u[0] = *reinterpret_cast<const uint4*>(pb[g] + k);
        fb.u[1] = *reinterpret_cast<const uint4*>(pb[g] + k + 16);
        acc[0][g] = __builtin_amdgcn_wmma_f32_16x16x32_bf16(
            false, fa0.v, false, fb.v, (short)0, acc[0][g], false, false);
        acc[1][g] = __builtin_amdgcn_wmma_f32_16x16x32_bf16(
            false, fa1.v, false, fb.v, (short)0, acc[1][g], false, false);
      }
    }
  }

  // ---- recurrent projection: A = h_{t-1}, B rows = U[g*H + hcol, :] ----
  {
    const __hip_bfloat16* a0 = hprev + (size_t)(mbase + lr) * H_ + hi * 8;
    const __hip_bfloat16* a1 = a0 + (size_t)16 * H_;
    const __hip_bfloat16* pb[4];
#pragma unroll
    for (int g = 0; g < 4; ++g)
      pb[g] = uB + ((size_t)g * H_ + hbase + lr) * H_ + hi * 8;

    for (int k = 0; k < H_; k += 32) {
      Frag fa0, fa1;
      fa0.u[0] = *reinterpret_cast<const uint4*>(a0 + k);
      fa0.u[1] = *reinterpret_cast<const uint4*>(a0 + k + 16);
      fa1.u[0] = *reinterpret_cast<const uint4*>(a1 + k);
      fa1.u[1] = *reinterpret_cast<const uint4*>(a1 + k + 16);
#pragma unroll
      for (int g = 0; g < 4; ++g) {
        Frag fb;
        fb.u[0] = *reinterpret_cast<const uint4*>(pb[g] + k);
        fb.u[1] = *reinterpret_cast<const uint4*>(pb[g] + k + 16);
        acc[0][g] = __builtin_amdgcn_wmma_f32_16x16x32_bf16(
            false, fa0.v, false, fb.v, (short)0, acc[0][g], false, false);
        acc[1][g] = __builtin_amdgcn_wmma_f32_16x16x32_bf16(
            false, fa1.v, false, fb.v, (short)0, acc[1][g], false, false);
      }
    }
  }

  // ---- gates + state update, in-register on the C fragments ----
  // C/D layout: VGPR j, lanes 0-15 -> M=j, lanes 16-31 -> M=j+8; N = lane%16.
  const int hc = hbase + lr;
  const float bF = bias[0 * H_ + hc];
  const float bI = bias[1 * H_ + hc];
  const float bO = bias[2 * H_ + hc];
  const float bG = bias[3 * H_ + hc];

#pragma unroll
  for (int s = 0; s < 2; ++s) {
#pragma unroll
    for (int j = 0; j < 8; ++j) {
      const int m = mbase + s * 16 + hi * 8 + j;
      const size_t idx = (size_t)m * H_ + hc;
      const float zf = acc[s][0][j] + bF;
      const float zi = acc[s][1][j] + bI;
      const float zo = acc[s][2][j] + bO;
      const float zg = acc[s][3][j] + bG;
      const float f = 1.f / (1.f + __expf(-zf));
      const float i = 1.f / (1.f + __expf(-zi));
      const float o = 1.f / (1.f + __expf(-zo));
      const float g = tanhf(zg);
      const float cn = i * g + f * cstate[idx];
      cstate[idx] = cn;
      const float h = o * tanhf(cn);
      hout[idx] = __float2bfloat16(h);
      if (houtf) houtf[idx] = h;
    }
  }
}

extern "C" void kernel_launch(void* const* d_in, const int* in_sizes, int n_in,
                              void* d_out, int out_size, void* d_ws, size_t ws_size,
                              hipStream_t stream) {
  const float* x  = (const float*)d_in[0];
  const float* w0 = (const float*)d_in[1];
  const float* b0 = (const float*)d_in[2];
  const float* u0 = (const float*)d_in[3];
  const float* w1 = (const float*)d_in[4];
  const float* b1 = (const float*)d_in[5];
  const float* u1 = (const float*)d_in[6];
  const float* w2 = (const float*)d_in[7];
  const float* b2 = (const float*)d_in[8];
  const float* u2 = (const float*)d_in[9];
  float* out = (float*)d_out;

  // ---- carve workspace (~233 MB total) ----
  size_t off = 0;
  auto carve = [&](size_t elems, size_t esize) -> void* {
    void* r = (char*)d_ws + off;
    off += (elems * esize + 255) & ~(size_t)255;
    return r;
  };
  __hip_bfloat16* w0b = (__hip_bfloat16*)carve((size_t)4 * H_ * D_, 2);
  __hip_bfloat16* u0b = (__hip_bfloat16*)carve((size_t)4 * H_ * H_, 2);
  __hip_bfloat16* w1b = (__hip_bfloat16*)carve((size_t)4 * H_ * H_, 2);
  __hip_bfloat16* u1b = (__hip_bfloat16*)carve((size_t)4 * H_ * H_, 2);
  __hip_bfloat16* w2b = (__hip_bfloat16*)carve((size_t)4 * H_ * H_, 2);
  __hip_bfloat16* u2b = (__hip_bfloat16*)carve((size_t)4 * H_ * H_, 2);
  __hip_bfloat16* xb  = (__hip_bfloat16*)carve((size_t)T_ * B_ * D_, 2);
  __hip_bfloat16* hsA = (__hip_bfloat16*)carve((size_t)T_ * B_ * H_, 2);  // L0 out
  __hip_bfloat16* hsB = (__hip_bfloat16*)carve((size_t)T_ * B_ * H_, 2);  // L1 out
  __hip_bfloat16* hsC = (__hip_bfloat16*)carve((size_t)2 * B_ * H_, 2);   // L2 ping-pong
  __hip_bfloat16* h0  = (__hip_bfloat16*)carve((size_t)B_ * H_, 2);
  float* c0 = (float*)carve((size_t)B_ * H_, 4);
  float* c1 = (float*)carve((size_t)B_ * H_, 4);
  float* c2 = (float*)carve((size_t)B_ * H_, 4);

  // ---- fp32 -> bf16 conversions (weights + layer-0 input) ----
  auto cvt = [&](const float* src, __hip_bfloat16* dst, long n) {
    cvt_f32_bf16<<<2048, 256, 0, stream>>>(src, dst, n);
  };
  cvt(x,  xb,  (long)T_ * B_ * D_);
  cvt(w0, w0b, (long)4 * H_ * D_);
  cvt(u0, u0b, (long)4 * H_ * H_);
  cvt(w1, w1b, (long)4 * H_ * H_);
  cvt(u1, u1b, (long)4 * H_ * H_);
  cvt(w2, w2b, (long)4 * H_ * H_);
  cvt(u2, u2b, (long)4 * H_ * H_);

  hipMemsetAsync(h0, 0, (size_t)B_ * H_ * 2, stream);  // h_{-1} = 0
  hipMemsetAsync(c0, 0, (size_t)B_ * H_ * 4, stream);  // c_{-1} = 0
  hipMemsetAsync(c1, 0, (size_t)B_ * H_ * 4, stream);
  hipMemsetAsync(c2, 0, (size_t)B_ * H_ * 4, stream);

  StepArgs a;
  a.xs[0] = xb;  a.xs[1] = hsA; a.xs[2] = hsB;
  a.w[0] = w0b;  a.w[1] = w1b;  a.w[2] = w2b;
  a.u[0] = u0b;  a.u[1] = u1b;  a.u[2] = u2b;
  a.bias[0] = b0; a.bias[1] = b1; a.bias[2] = b2;
  a.c[0] = c0;   a.c[1] = c1;   a.c[2] = c2;
  a.hs[0] = hsA; a.hs[1] = hsB; a.hs[2] = hsC;
  a.f32 = out;
  a.h0 = h0;
  a.din[0] = D_; a.din[1] = H_; a.din[2] = H_;
  a.tmask[0] = 0x7fffffff; a.tmask[1] = 0x7fffffff; a.tmask[2] = 1;

  // One launch per diagonal: (L0,d), (L1,d-1), (L2,d-2) run concurrently.
  // Intra-launch buffers are disjoint: L0 writes hsA[d], L1 reads hsA[d-1]
  // and writes hsB[d-1], L2 reads hsB[d-2] and writes hsC ping-pong + out.
  for (int d = 0; d < T_ + 2; ++d) {
    a.d = d;
    lstm_wavefront<<<192, 128, 0, stream>>>(a);
  }
}
